// MMOE_87686052315222
// MI455X (gfx1250) — matrix operand order repaired and checked by reading, not compiled, not run on hardware
//
#include <hip/hip_runtime.h>

typedef __attribute__((ext_vector_type(16))) __bf16 v16bf;
typedef __attribute__((ext_vector_type(8)))  float  v8f;

#define PX 264   // bf16 LDS row pitch (256 + 8 pad) -> 528B rows, conflict-friendly
#define PT 132   // fp32 LDS row pitch for tower staging (same 528B rows)

struct Q2 { uint4 a, b; };

__device__ __forceinline__ unsigned short f2bf(float f) {
  unsigned u = __float_as_uint(f);
  unsigned r = u + 0x7FFFu + ((u >> 16) & 1u);   // round-to-nearest-even
  return (unsigned short)(r >> 16);
}

__device__ __forceinline__ v8f vzero8() {
  v8f z = {0.f,0.f,0.f,0.f,0.f,0.f,0.f,0.f};
  return z;
}

__device__ __forceinline__ v8f wmma_bf16(v16bf a, v16bf b, v8f c) {
  return __builtin_amdgcn_wmma_f32_16x16x32_bf16(false, a, false, b, (short)0, c, false, false);
}

// A fragment (16x32 bf16) from LDS: lane = h*16 + r ; row = r, elems {k0+h*8 .. +7} and {k0+16+h*8 .. +7}
__device__ __forceinline__ v16bf ld_A(const unsigned short* sh, int row, int k0h) {
  const unsigned short* p = sh + row * PX + k0h;
  Q2 q;
  q.a = *(const uint4*)p;
  q.b = *(const uint4*)(p + 16);
  return __builtin_bit_cast(v16bf, q);
}

// B fragment (32x16 bf16), pre-packed fragment-contiguous: lane's 16 values at frag + lane*16
__device__ __forceinline__ v16bf ld_B(const unsigned short* frag, int lane) {
  const unsigned short* p = frag + lane * 16;
  Q2 q;
  q.a = *(const uint4*)p;
  q.b = *(const uint4*)(p + 8);
  return __builtin_bit_cast(v16bf, q);
}

// ---------------- weight pre-pack kernels ----------------
// Fragment order: [mat][ntile][ktile][lane][j], 512 bf16 per fragment.
// Element: k = kt*32 + (lane>>4)*8 + (j&7) + (j>>3)*16 ; n = nt*16 + (lane&15); src is row-major [K][N].
__global__ void pack_frags(const float* __restrict__ src, unsigned short* __restrict__ dst,
                           int nmat, int K, int N) {
  int total = nmat * K * N;
  int stride = gridDim.x * blockDim.x;
  for (int idx = blockIdx.x * blockDim.x + threadIdx.x; idx < total; idx += stride) {
    int per = K * N;
    int mat = idx / per, r = idx - mat * per;
    int f = r >> 9, lj = r & 511;
    int lane = lj >> 4, j = lj & 15;
    int ktC = K >> 5;
    int nt = f / ktC, kt = f - nt * ktC;
    int k = kt * 32 + ((lane >> 4) << 3) + (j & 7) + ((j >> 3) << 4);
    int n = nt * 16 + (lane & 15);
    dst[idx] = f2bf(src[mat * per + k * N + n]);
  }
}

// gW is [T][IN][E]; pack into one 256x16 operand with n = t*8+e  (8 k-tiles)
__global__ void pack_gw(const float* __restrict__ gW, unsigned short* __restrict__ dst) {
  int idx = blockIdx.x * blockDim.x + threadIdx.x;   // 8*512 = 4096
  if (idx < 4096) {
    int kt = idx >> 9, lj = idx & 511;
    int lane = lj >> 4, j = lj & 15;
    int k = kt * 32 + ((lane >> 4) << 3) + (j & 7) + ((j >> 3) << 4);
    int n = lane & 15;
    int t = n >> 3, e = n & 7;
    dst[idx] = f2bf(gW[(t * 256 + k) * 8 + e]);
  }
}

// ---------------- fused MMoE forward ----------------
__global__ __launch_bounds__(256, 1) void mmoe_fused(
    const int* __restrict__ uidx, const int* __restrict__ iidx,
    const float* __restrict__ utab, const float* __restrict__ itab,
    const unsigned short* __restrict__ w1p, const float* __restrict__ eb1,
    const unsigned short* __restrict__ w2p, const float* __restrict__ eb2,
    const unsigned short* __restrict__ gwp, const float* __restrict__ gb,
    const unsigned short* __restrict__ tw1p, const float* __restrict__ tb1,
    const float* __restrict__ tw2, const float* __restrict__ tb2,
    float* __restrict__ out, int Bn)
{
  __shared__ unsigned short xsh[128 * PX];  // x bf16  -> reused as tower-hidden fp32 [128][PT]
  __shared__ unsigned short hsh[128 * PX];  // h1 bf16 -> reused as mmoe bf16 staging
  __shared__ float gsh[2 * 128 * 8];        // gate weights [t][row][e]

  const int tid  = threadIdx.x;
  const int lane = tid & 31;
  const int wid  = tid >> 5;
  const int r15  = lane & 15;
  const int hh   = lane >> 4;
  const int R    = wid << 4;                 // local row base of this wave's 16-row tile
  const int rowBase = blockIdx.x * 128;

  // ---- Phase 0: gather embeddings, convert to bf16 in LDS ----
  for (int i = tid; i < 128 * 64; i += 256) {
    int row = i >> 6, c4 = i & 63;
    float4 v = (c4 < 32)
      ? *(const float4*)(utab + (size_t)uidx[rowBase + row] * 128 + c4 * 4)
      : *(const float4*)(itab + (size_t)iidx[rowBase + row] * 128 + (c4 - 32) * 4);
    ushort4 u;
    u.x = f2bf(v.x); u.y = f2bf(v.y); u.z = f2bf(v.z); u.w = f2bf(v.w);
    *(ushort4*)(xsh + row * PX + c4 * 4) = u;
  }
  __syncthreads();

  // Persistent A-fragments of x for this wave's tile (K = 256 -> 8 k-tiles)
  v16bf ax[8];
#pragma unroll
  for (int kt = 0; kt < 8; ++kt) ax[kt] = ld_A(xsh, R + r15, kt * 32 + hh * 8);

  // ---- Gates via WMMA: logits [16 rows x 16 (t*8+e)] ----
  {
    v8f gl = vzero8();
#pragma unroll
    for (int kt = 0; kt < 8; ++kt)
      gl = wmma_bf16(ax[kt], ld_B(gwp + kt * 512, lane), gl);
    int n = r15, t = n >> 3, e = n & 7;
    float gbn = gb[n];
#pragma unroll
    for (int v = 0; v < 8; ++v) {
      int row = R + v + hh * 8;
      gsh[(t * 128 + row) * 8 + e] = gl[v] + gbn;
    }
  }
  __syncthreads();
  // softmax over experts; one (t,row) per thread
  {
    int t = tid >> 7, row = tid & 127;
    float* g = &gsh[(t * 128 + row) * 8];
    float m = g[0];
#pragma unroll
    for (int e = 1; e < 8; ++e) m = fmaxf(m, g[e]);
    float s = 0.f, ev[8];
#pragma unroll
    for (int e = 0; e < 8; ++e) { ev[e] = __expf(g[e] - m); s += ev[e]; }
    float inv = 1.f / s;
#pragma unroll
    for (int e = 0; e < 8; ++e) g[e] = ev[e] * inv;
  }
  __syncthreads();

  // ---- Expert loop with gated accumulation ----
  v8f macc[2][16];
#pragma unroll
  for (int t = 0; t < 2; ++t)
#pragma unroll
    for (int nt = 0; nt < 16; ++nt) macc[t][nt] = vzero8();

  for (int e = 0; e < 8; ++e) {
    float garr[2][8];
#pragma unroll
    for (int t = 0; t < 2; ++t)
#pragma unroll
      for (int v = 0; v < 8; ++v)
        garr[t][v] = gsh[(t * 128 + R + v + hh * 8) * 8 + e];

    // GEMM1: h1 = relu(x @ eW1[e] + eb1[e])  -> bf16 LDS (own rows only)
#pragma unroll
    for (int nt = 0; nt < 16; ++nt) {
      v8f acc = vzero8();
      const unsigned short* bp = w1p + (size_t)((e * 16 + nt) * 8) * 512;
#pragma unroll
      for (int kt = 0; kt < 8; ++kt)
        acc = wmma_bf16(ax[kt], ld_B(bp + kt * 512, lane), acc);
      float bias = eb1[e * 256 + nt * 16 + r15];
#pragma unroll
      for (int v = 0; v < 8; ++v)
        hsh[(R + v + hh * 8) * PX + nt * 16 + r15] = f2bf(fmaxf(acc[v] + bias, 0.f));
    }

    // GEMM2: expert_out = relu(h1 @ eW2[e] + eb2[e]); mmoe[t] += g[t,e] * expert_out
    v16bf ah[8];
#pragma unroll
    for (int kt = 0; kt < 8; ++kt) ah[kt] = ld_A(hsh, R + r15, kt * 32 + hh * 8);
#pragma unroll
    for (int nt = 0; nt < 16; ++nt) {
      v8f acc = vzero8();
      const unsigned short* bp = w2p + (size_t)((e * 16 + nt) * 8) * 512;
#pragma unroll
      for (int kt = 0; kt < 8; ++kt)
        acc = wmma_bf16(ah[kt], ld_B(bp + kt * 512, lane), acc);
      float bias = eb2[e * 256 + nt * 16 + r15];
#pragma unroll
      for (int v = 0; v < 8; ++v) {
        float ov = fmaxf(acc[v] + bias, 0.f);
        macc[0][nt][v] += garr[0][v] * ov;
        macc[1][nt][v] += garr[1][v] * ov;
      }
    }
  }

  // ---- Towers (per task): th = relu(mmoe @ tW1 + tb1); out = th @ tW2 + tb2 ----
  float* thsh = (float*)xsh;  // reuse x region, same 528B row pitch, own rows only
#pragma unroll
  for (int t = 0; t < 2; ++t) {
#pragma unroll
    for (int nt = 0; nt < 16; ++nt)
#pragma unroll
      for (int v = 0; v < 8; ++v)
        hsh[(R + v + hh * 8) * PX + nt * 16 + r15] = f2bf(macc[t][nt][v]);

    v16bf am[8];
#pragma unroll
    for (int kt = 0; kt < 8; ++kt) am[kt] = ld_A(hsh, R + r15, kt * 32 + hh * 8);

#pragma unroll
    for (int nt = 0; nt < 8; ++nt) {
      v8f acc = vzero8();
      const unsigned short* bp = tw1p + (size_t)((t * 8 + nt) * 8) * 512;
#pragma unroll
      for (int kt = 0; kt < 8; ++kt)
        acc = wmma_bf16(am[kt], ld_B(bp + kt * 512, lane), acc);
      float bias = tb1[t * 128 + nt * 16 + r15];
#pragma unroll
      for (int v = 0; v < 8; ++v)
        thsh[(R + v + hh * 8) * PT + nt * 16 + r15] = fmaxf(acc[v] + bias, 0.f);
    }

    // final scalar head: each half-lane sums 64 cols, combine across lane^16
    float s = 0.f;
    const float* tr = thsh + (R + r15) * PT + hh * 64;
    const float* w  = tw2 + t * 128 + hh * 64;
#pragma unroll 8
    for (int i = 0; i < 64; ++i) s += tr[i] * w[i];
    s += __shfl_xor(s, 16);
    if (hh == 0) out[t * Bn + rowBase + R + r15] = s + tb2[t];
  }
}

// ---------------- host entry ----------------
extern "C" void kernel_launch(void* const* d_in, const int* in_sizes, int n_in,
                              void* d_out, int out_size, void* d_ws, size_t ws_size,
                              hipStream_t stream) {
  const int*   uidx = (const int*)d_in[0];
  const int*   iidx = (const int*)d_in[1];
  const float* utab = (const float*)d_in[2];
  const float* itab = (const float*)d_in[3];
  const float* eW1  = (const float*)d_in[4];
  const float* eb1  = (const float*)d_in[5];
  const float* eW2  = (const float*)d_in[6];
  const float* eb2  = (const float*)d_in[7];
  const float* gW   = (const float*)d_in[8];
  const float* gb   = (const float*)d_in[9];
  const float* tW1  = (const float*)d_in[10];
  const float* tb1  = (const float*)d_in[11];
  const float* tW2  = (const float*)d_in[12];
  const float* tb2  = (const float*)d_in[13];
  float* out = (float*)d_out;

  char* ws = (char*)d_ws;
  unsigned short* w1p  = (unsigned short*)(ws);                       // 1 MB
  unsigned short* w2p  = (unsigned short*)(ws + 1048576);             // 1 MB
  unsigned short* tw1p = (unsigned short*)(ws + 2097152);             // 128 KB
  unsigned short* gwp  = (unsigned short*)(ws + 2228224);             // 8 KB

  pack_frags<<<512, 256, 0, stream>>>(eW1, w1p, 8, 256, 256);
  pack_frags<<<512, 256, 0, stream>>>(eW2, w2p, 8, 256, 256);
  pack_frags<<<64, 256, 0, stream>>>(tW1, tw1p, 2, 256, 128);
  pack_gw<<<16, 256, 0, stream>>>(gW, gwp);

  int Bn = in_sizes[0];
  mmoe_fused<<<Bn / 128, 256, 0, stream>>>(uidx, iidx, utab, itab,
                                           w1p, eb1, w2p, eb2, gwp, gb,
                                           tw1p, tb1, tW2, tb2, out, Bn);
}
